// NoisyTopKRouter_44427141710497
// MI455X (gfx1250) — matrix operand order repaired and checked by reading, not compiled, not run on hardware
//
#include <hip/hip_runtime.h>

typedef __attribute__((ext_vector_type(16))) __bf16 v16bf;
typedef __attribute__((ext_vector_type(8)))  float  v8f;

#define N_EMBD     2048
#define N_EXP      64
#define NCOLS      128               // route(64) || noise(64)
#define TOK_PER_WG 32                // two 16-row M-tiles per wave, shared B
#define KSTEP      32                // K per wmma_f32_16x16x32_bf16
#define NKB        (N_EMBD / KSTEP)  // 64
#define A_PAD      8
#define A_STRIDE   (N_EMBD + A_PAD)  // 2056 bf16 elements per staged row

__device__ __forceinline__ unsigned short f2bf(float f) {
    // round-to-nearest-even f32 -> bf16
    unsigned u = __float_as_uint(f);
    unsigned r = (u + 0x7FFFu + ((u >> 16) & 1u)) >> 16;
    return (unsigned short)r;
}

// ---------------------------------------------------------------------------
// Pre-pack W_route||W_noise (f32, [2048 x 64] each, expert-contiguous) into
// bf16 in this kernel's B-operand order (K contiguous per lane half):
//   g = ((ct*64 + kb)*32 + lane), each g owns 16 bf16 (32 bytes).
//   lane 0..15  -> column N = ct*16 + lane, K = kb*32 + 0..15
//   lane 16..31 -> column N = ct*16 + lane-16, K = kb*32 + 16..31
// ---------------------------------------------------------------------------
__global__ void pack_w_kernel(const float* __restrict__ Wr,
                              const float* __restrict__ Wn,
                              unsigned short* __restrict__ Bp) {
    int g    = blockIdx.x * blockDim.x + threadIdx.x;   // 0 .. 8*64*32-1
    int lane = g & 31;
    int kb   = (g >> 5) & (NKB - 1);
    int ct   = g >> 11;                                 // 0..7 column tile
    int col  = ct * 16 + (lane & 15);
    int kbase = kb * KSTEP + (lane >> 4) * 16;
    const float* W = (col < N_EXP) ? Wr : Wn;
    int c = col & (N_EXP - 1);
    union { unsigned short s[16]; uint4 q[2]; } t;
#pragma unroll
    for (int e = 0; e < 16; ++e)
        t.s[e] = f2bf(W[(size_t)(kbase + e) * N_EXP + c]);
    uint4* dst = (uint4*)(Bp + (size_t)g * 16);
    dst[0] = t.q[0];
    dst[1] = t.q[1];
}

// ---------------------------------------------------------------------------
// Fused router: dual GEMM (bf16 WMMA, f32 accum) + noisy top-2 + softmax +
// scatter. One block = 32 tokens, 8 waves; wave ct owns combined columns
// [ct*16, ct*16+16) and computes TWO 16x16 C tiles (token rows 0-15 and
// 16-31) that share each B fetch -> halves L2 weight traffic.
// ---------------------------------------------------------------------------
__global__ void __launch_bounds__(256)
router_kernel(const float* __restrict__ x,
              const float* __restrict__ eps,
              const float* __restrict__ br,
              const float* __restrict__ bn,
              const unsigned short* __restrict__ Bp,
              float* __restrict__ rout,
              int* __restrict__ idxout) {
    __shared__ __align__(16) unsigned short sA[TOK_PER_WG * A_STRIDE]; // ~128KB
    __shared__ float sL[TOK_PER_WG * NCOLS];                            // 16KB
    __shared__ float sP[TOK_PER_WG * 2];
    __shared__ int   sI[TOK_PER_WG * 2];

    const int tid  = threadIdx.x;
    const int tok0 = blockIdx.x * TOK_PER_WG;

    // ---- stage 32 x-rows into LDS as bf16 (coalesced float2 loads) ----
    const float2* xsrc = (const float2*)(x + (size_t)tok0 * N_EMBD);
    for (int i = tid; i < TOK_PER_WG * (N_EMBD / 2); i += 256) {
        int trow = i >> 10;          // /(N_EMBD/2)
        int kp   = i & (N_EMBD / 2 - 1);
        float2 v = xsrc[(size_t)trow * (N_EMBD / 2) + kp];
        unsigned pk = (unsigned)f2bf(v.x) | ((unsigned)f2bf(v.y) << 16);
        *(unsigned*)&sA[trow * A_STRIDE + kp * 2] = pk;
    }
    __syncthreads();

    // ---- dual GEMM via WMMA, 2 M-tiles per wave ----
    const int wave = tid >> 5;
    const int lane = tid & 31;
    const int half = lane >> 4;      // K-half select (contiguous K order used here)
    const int nloc = lane & 15;
    const int col  = wave * 16 + nloc;
    const float bias = (col < N_EXP) ? br[col] : bn[col - N_EXP];

    // Loop-invariant LDS base pointers (keeps the 16*A_STRIDE delta out of
    // the loop so DS loads use small immediate offsets only).
    const unsigned short* pA0 = sA + nloc * A_STRIDE + half * 16;  // rows 0-15
    const unsigned short* pA1 = pA0 + 16 * A_STRIDE;               // rows 16-31

    v8f acc0 = {};
    v8f acc1 = {};
    const uint4* bq = (const uint4*)Bp + ((size_t)(wave * NKB) * 32 + lane) * 2;
#pragma unroll 8
    for (int kb = 0; kb < NKB; ++kb) {
        // A: lane = token row (M = lane%16); K = k0+half*16 .. +16 contiguous,
        //    matching the packed B order.
        const int off = kb * KSTEP;          // elements; *2 bytes <= 4064
        union { v16bf v; uint4 q[2]; } A0, A1, B;
        A0.q[0] = *(const uint4*)(pA0 + off);
        A0.q[1] = *(const uint4*)(pA0 + off + 8);
        A1.q[0] = *(const uint4*)(pA1 + off);
        A1.q[1] = *(const uint4*)(pA1 + off + 8);
        B.q[0] = bq[0];
        B.q[1] = bq[1];
        bq += 64;                    // next kb block: 32 lanes * 2 uint4
        acc0 = __builtin_amdgcn_wmma_f32_16x16x32_bf16(
                   false, A0.v, false, B.v, (short)0, acc0, false, false);
        acc1 = __builtin_amdgcn_wmma_f32_16x16x32_bf16(
                   false, A1.v, false, B.v, (short)0, acc1, false, false);
    }
#pragma unroll
    for (int r = 0; r < 8; ++r) {
        int trow = r + 8 * half;     // C-layout: VGPR r -> M = r (+8 for hi lanes)
        sL[trow * NCOLS + col]        = acc0[r] + bias;
        sL[(trow + 16) * NCOLS + col] = acc1[r] + bias;
    }
    __syncthreads();

    // ---- noisy top-2 + softmax (one lane per token) ----
    if (tid < TOK_PER_WG) {
        int trow = tid;
        const float* erow = eps + (size_t)(tok0 + trow) * N_EXP;
        float m0 = -3.0e38f, m1 = -3.0e38f;
        int   i0 = 0, i1 = 0;
        for (int e = 0; e < N_EXP; ++e) {
            float ra = sL[trow * NCOLS + e];
            float nr = sL[trow * NCOLS + N_EXP + e];
            float sp = (nr > 0.f) ? (nr + log1pf(__expf(-nr)))
                                  : log1pf(__expf(nr));
            float v = ra + erow[e] * sp;
            if (v > m0)      { m1 = m0; i1 = i0; m0 = v; i0 = e; }
            else if (v > m1) { m1 = v; i1 = e; }
        }
        float p1 = 1.f / (1.f + __expf(m0 - m1));   // softmax over {m0, m1}
        float p0 = 1.f - p1;
        sP[trow * 2] = p0;  sP[trow * 2 + 1] = p1;
        sI[trow * 2] = i0;  sI[trow * 2 + 1] = i1;
        idxout[(size_t)(tok0 + trow) * 2]     = i0;
        idxout[(size_t)(tok0 + trow) * 2 + 1] = i1;
    }
    __syncthreads();

    // ---- scatter probabilities (coalesced) ----
    for (int i = tid; i < TOK_PER_WG * N_EXP; i += 256) {
        int trow = i >> 6;
        int e    = i & (N_EXP - 1);
        float v = (e == sI[trow * 2])     ? sP[trow * 2]
                : (e == sI[trow * 2 + 1]) ? sP[trow * 2 + 1] : 0.f;
        rout[(size_t)tok0 * N_EXP + i] = v;
    }
}

extern "C" void kernel_launch(void* const* d_in, const int* in_sizes, int n_in,
                              void* d_out, int out_size, void* d_ws, size_t ws_size,
                              hipStream_t stream) {
    (void)n_in; (void)out_size; (void)ws_size;
    const float* x  = (const float*)d_in[0];
    const float* ep = (const float*)d_in[1];
    const float* Wr = (const float*)d_in[2];
    const float* br = (const float*)d_in[3];
    const float* Wn = (const float*)d_in[4];
    const float* bn = (const float*)d_in[5];

    const int tokens = in_sizes[0] / N_EMBD;     // B*S
    unsigned short* Bp = (unsigned short*)d_ws;  // 512 KB packed bf16 weights
    float* rout  = (float*)d_out;                // [tokens, 64] f32
    int*   idxou = (int*)(rout + (size_t)tokens * N_EXP); // [tokens, 2] i32

    pack_w_kernel<<<(8 * NKB * 32) / 256, 256, 0, stream>>>(Wr, Wn, Bp);
    router_kernel<<<tokens / TOK_PER_WG, 256, 0, stream>>>(x, ep, br, bn, Bp,
                                                           rout, idxou);
}
